// RGCN_83176336654883
// MI455X (gfx1250) — matrix operand order repaired and checked by reading, not compiled, not run on hardware
//
#include <hip/hip_runtime.h>
#include <hip/hip_bf16.h>

// ---------------------------------------------------------------------------
// RGCN (2-layer) for MI455X / gfx1250, wave32 + WMMA bf16 hi/lo split GEMMs.
//   layer: out = x@root + b + sum_r scatter_mean_r( (x@W[r])[src] -> dst )
// Strategy: per-relation dense transform H_r = x@W[r] (25.6MB, L2-resident),
// then edge scatter (float4-vectorized gather + f32 atomics) with
// precomputed 1/max(cnt_r[dst],1) scaling so all relations accumulate into
// one buffer.
// ---------------------------------------------------------------------------

#define DIN 64
#define XPITCH 72  // 64 + 8 bf16 pad -> 144B row pitch, conflict-free col reads

typedef __bf16 bh8   __attribute__((ext_vector_type(8)));
typedef __bf16 v16bf __attribute__((ext_vector_type(16)));
typedef float  v8f   __attribute__((ext_vector_type(8)));

static __device__ __forceinline__ v16bf cat8(bh8 a, bh8 b) {
    return __builtin_shufflevector(a, b, 0,1,2,3,4,5,6,7,8,9,10,11,12,13,14,15);
}

static __device__ __forceinline__ void split_bf16(float v, __bf16& hi, __bf16& lo) {
    hi = (__bf16)v;
    lo = (__bf16)(v - (float)hi);
}

// ---------------------------------------------------------------------------
// GEMM: Out[n, 0:DOUT] = X[n, 0:64] @ W(64 x DOUT) (+bias), optional relu(X).
// Block = 128 threads (4 waves), tile = 64 nodes x DOUT cols, K = 64 in LDS.
// bf16 split: acc += Ahi*Bhi + Ahi*Blo + Alo*Bhi  (fp32 accumulate).
// ---------------------------------------------------------------------------
template<int DOUT, bool RELU, bool BIAS>
__global__ __launch_bounds__(128)
void rgcn83_gemm(const float* __restrict__ X,
                 const float* __restrict__ W,
                 const float* __restrict__ bias,
                 float* __restrict__ Out,
                 int N)
{
    __shared__ __align__(16) __bf16 sXhi[64][XPITCH];
    __shared__ __align__(16) __bf16 sXlo[64][XPITCH];
    __shared__ __align__(16) __bf16 sWhi[DOUT][XPITCH];  // transposed: [col][k]
    __shared__ __align__(16) __bf16 sWlo[DOUT][XPITCH];

    const int tid = threadIdx.x;
    const int n0  = blockIdx.x * 64;

    // ---- load X tile (64 x 64 f32), split into bf16 hi/lo planes ----
    for (int idx = tid; idx < 64 * 16; idx += 128) {
        const int r  = idx >> 4;
        const int c4 = (idx & 15) << 2;
        float4 v = make_float4(0.f, 0.f, 0.f, 0.f);
        const int row = n0 + r;
        if (row < N) v = *(const float4*)(X + (size_t)row * DIN + c4);
        float f[4] = {v.x, v.y, v.z, v.w};
        #pragma unroll
        for (int u = 0; u < 4; ++u) {
            float t = RELU ? fmaxf(f[u], 0.f) : f[u];
            __bf16 hi, lo; split_bf16(t, hi, lo);
            sXhi[r][c4 + u] = hi;
            sXlo[r][c4 + u] = lo;
        }
    }
    // ---- load W (64 x DOUT f32) transposed into LDS as [col][k] ----
    for (int idx = tid; idx < 64 * (DOUT / 4); idx += 128) {
        const int r  = idx / (DOUT / 4);
        const int c4 = (idx % (DOUT / 4)) * 4;
        float4 v = *(const float4*)(W + (size_t)r * DOUT + c4);
        float f[4] = {v.x, v.y, v.z, v.w};
        #pragma unroll
        for (int u = 0; u < 4; ++u) {
            __bf16 hi, lo; split_bf16(f[u], hi, lo);
            sWhi[c4 + u][r] = hi;
            sWlo[c4 + u][r] = lo;
        }
    }
    __syncthreads();

    const int wave    = tid >> 5;
    const int lane    = tid & 31;
    const int h       = lane >> 4;   // half-wave index
    const int lrow    = lane & 15;
    const int rowbase = wave * 16;

    constexpr int NT = DOUT / 16;
    v8f acc[NT] = {};

    #pragma unroll
    for (int k0 = 0; k0 < 64; k0 += 32) {
        // A (16x32 bf16): lane holds row=lrow, K = k0 + {8h..8h+7, 16+8h..23+8h}
        bh8 xh0 = *(const bh8*)&sXhi[rowbase + lrow][k0 + 8 * h];
        bh8 xh1 = *(const bh8*)&sXhi[rowbase + lrow][k0 + 16 + 8 * h];
        bh8 xl0 = *(const bh8*)&sXlo[rowbase + lrow][k0 + 8 * h];
        bh8 xl1 = *(const bh8*)&sXlo[rowbase + lrow][k0 + 16 + 8 * h];
        v16bf ahi = cat8(xh0, xh1);
        v16bf alo = cat8(xl0, xl1);

        #pragma unroll
        for (int ct = 0; ct < NT; ++ct) {
            // B (32x16 bf16): lane holds col=lrow, K = k0 + {8h..8h+7, 16+8h..23+8h}
            bh8 wh0 = *(const bh8*)&sWhi[ct * 16 + lrow][k0 + 8 * h];
            bh8 wh1 = *(const bh8*)&sWhi[ct * 16 + lrow][k0 + 16 + 8 * h];
            bh8 wl0 = *(const bh8*)&sWlo[ct * 16 + lrow][k0 + 8 * h];
            bh8 wl1 = *(const bh8*)&sWlo[ct * 16 + lrow][k0 + 16 + 8 * h];
            v16bf bhi = cat8(wh0, wh1);
            v16bf blo = cat8(wl0, wl1);
            acc[ct] = __builtin_amdgcn_wmma_f32_16x16x32_bf16(
                false, ahi, false, bhi, (short)0, acc[ct], false, false);
            acc[ct] = __builtin_amdgcn_wmma_f32_16x16x32_bf16(
                false, ahi, false, blo, (short)0, acc[ct], false, false);
            acc[ct] = __builtin_amdgcn_wmma_f32_16x16x32_bf16(
                false, alo, false, bhi, (short)0, acc[ct], false, false);
        }
    }

    // ---- store C/D: VGPR j -> M = j + 8h, N = lrow ----
    #pragma unroll
    for (int ct = 0; ct < NT; ++ct) {
        const int col = ct * 16 + lrow;
        const float bv = BIAS ? bias[col] : 0.f;
        #pragma unroll
        for (int j = 0; j < 8; ++j) {
            const int row = n0 + rowbase + j + 8 * h;
            if (row < N) Out[(size_t)row * DOUT + col] = acc[ct][j] + bv;
        }
    }
}

// ---------------------------------------------------------------------------
// Edge kernels
// ---------------------------------------------------------------------------
__global__ void rgcn83_zero(float* __restrict__ p, int n) {
    int i = blockIdx.x * blockDim.x + threadIdx.x;
    if (i < n) p[i] = 0.f;
}

__global__ void rgcn83_count(const int* __restrict__ et, const int* __restrict__ dst,
                             float* __restrict__ cnt, int E, int N) {
    int e = blockIdx.x * blockDim.x + threadIdx.x;
    if (e < E) atomicAdd(cnt + (size_t)et[e] * N + dst[e], 1.f);
}

__global__ void rgcn83_recip(float* __restrict__ p, int n) {
    int i = blockIdx.x * blockDim.x + threadIdx.x;
    if (i < n) p[i] = 1.f / fmaxf(p[i], 1.f);
}

// One relation per pass. DOUT/4 threads per edge, each moving a float4:
// gather becomes global_load_b128 and edge-index reads are amortized 4x.
// Edge id is uniform across groups of DOUT/4 lanes, so the r!=rel early-out
// stays divergence-free at sub-wave granularity.
template<int DOUT>
__global__ void rgcn83_scatter(const float* __restrict__ H,
                               const float* __restrict__ inv,
                               const int* __restrict__ src,
                               const int* __restrict__ dst,
                               const int* __restrict__ et,
                               float* __restrict__ out,
                               int rel, int E, int N)
{
    constexpr int TPE = DOUT / 4;  // threads per edge
    long long g = (long long)blockIdx.x * blockDim.x + threadIdx.x;
    int e  = (int)(g / TPE);
    int j4 = (int)(g % TPE) * 4;
    if (e >= E) return;
    if (et[e] != rel) return;
    const int d = dst[e];
    const float w = inv[(size_t)rel * N + d];
    const float4 hv = *(const float4*)(H + (size_t)src[e] * DOUT + j4);
    float* o = out + (size_t)d * DOUT + j4;
    atomicAdd(o + 0, hv.x * w);
    atomicAdd(o + 1, hv.y * w);
    atomicAdd(o + 2, hv.z * w);
    atomicAdd(o + 3, hv.w * w);
}

// ---------------------------------------------------------------------------
// Host launch
// ---------------------------------------------------------------------------
static inline int cdiv64(long long a, long long b) { return (int)((a + b - 1) / b); }

extern "C" void kernel_launch(void* const* d_in, const int* in_sizes, int n_in,
                              void* d_out, int out_size, void* d_ws, size_t ws_size,
                              hipStream_t stream) {
    const float* x     = (const float*)d_in[0];
    const float* W1    = (const float*)d_in[1];
    const float* root1 = (const float*)d_in[2];
    const float* b1    = (const float*)d_in[3];
    const float* W2    = (const float*)d_in[4];
    const float* root2 = (const float*)d_in[5];
    const float* b2    = (const float*)d_in[6];
    const int*   src   = (const int*)d_in[7];
    const int*   dst   = (const int*)d_in[8];
    const int*   et    = (const int*)d_in[9];

    const int N = in_sizes[0] / DIN;   // 100000
    const int E = in_sizes[7];         // 1250000
    const int R = 8;

    float* inv  = (float*)d_ws;                 // 8*N
    float* out1 = inv  + (size_t)R * N;         // N*64
    float* Hbuf = out1 + (size_t)N * 64;        // N*64 (reused both layers)
    float* out  = (float*)d_out;                // N*32

    // topology-only reciprocal mean weights (shared by both layers)
    rgcn83_zero <<<cdiv64((long long)R * N, 256), 256, 0, stream>>>(inv, R * N);
    rgcn83_count<<<cdiv64(E, 256), 256, 0, stream>>>(et, dst, inv, E, N);
    rgcn83_recip<<<cdiv64((long long)R * N, 256), 256, 0, stream>>>(inv, R * N);

    const int gN = cdiv64(N, 64);

    // ---- layer 1: out1 = x@root1 + b1 + sum_r mean_r ----
    rgcn83_gemm<64, false, true ><<<gN, 128, 0, stream>>>(x, root1, b1, out1, N);
    for (int r = 0; r < R; ++r) {
        rgcn83_gemm<64, false, false><<<gN, 128, 0, stream>>>(
            x, W1 + (size_t)r * 64 * 64, nullptr, Hbuf, N);
        rgcn83_scatter<64><<<cdiv64((long long)E * 16, 256), 256, 0, stream>>>(
            Hbuf, inv, src, dst, et, out1, r, E, N);
    }

    // ---- layer 2 (relu folded into GEMM loads): out = relu(out1)@root2+b2 + ... ----
    rgcn83_gemm<32, true, true ><<<gN, 128, 0, stream>>>(out1, root2, b2, out, N);
    for (int r = 0; r < R; ++r) {
        rgcn83_gemm<32, true, false><<<gN, 128, 0, stream>>>(
            out1, W2 + (size_t)r * 64 * 32, nullptr, Hbuf, N);
        rgcn83_scatter<32><<<cdiv64((long long)E * 8, 256), 256, 0, stream>>>(
            Hbuf, inv, src, dst, et, out, r, E, N);
    }
}
